// GraphSAGE_16501264351517
// MI455X (gfx1250) — matrix operand order repaired and checked by reading, not compiled, not run on hardware
//
#include <hip/hip_runtime.h>
#include <hip/hip_bf16.h>
#include <math.h>

typedef __attribute__((ext_vector_type(16))) __bf16 v16bf;
typedef __attribute__((ext_vector_type(8)))  float  v8f;

#define N_NODES 100000
#define N_EDGES 1600000
#define FEATS   128

// ---------------------------------------------------------------------------
// Degree: deg[dst[e]] += 1
// ---------------------------------------------------------------------------
__global__ void deg_kernel(const int* __restrict__ dst, float* __restrict__ deg, int nE) {
    int e = blockIdx.x * blockDim.x + threadIdx.x;
    if (e < nE) unsafeAtomicAdd(deg + dst[e], 1.0f);
}

__global__ void inv_kernel(float* __restrict__ d, int n) {
    int i = blockIdx.x * blockDim.x + threadIdx.x;
    if (i < n) d[i] = 1.0f / fmaxf(d[i], 1.0f);
}

// ---------------------------------------------------------------------------
// Scatter-add: agg[dst[e]][:] += h[src[e]][:]   (4 floats per thread)
// ---------------------------------------------------------------------------
__global__ void scatter_add(const float* __restrict__ h, const int* __restrict__ src,
                            const int* __restrict__ dst, float* __restrict__ agg, int nE) {
    int t = blockIdx.x * blockDim.x + threadIdx.x;
    int e = t >> 5;
    if (e >= nE) return;
    int c = (t & 31) * 4;
    int s = src[e], d = dst[e];
    const float4 v = *(const float4*)(h + (size_t)s * FEATS + c);
    float* o = agg + (size_t)d * FEATS + c;
    unsafeAtomicAdd(o + 0, v.x);
    unsafeAtomicAdd(o + 1, v.y);
    unsafeAtomicAdd(o + 2, v.z);
    unsafeAtomicAdd(o + 3, v.w);
}

// ---------------------------------------------------------------------------
// Pack fp32 weight [K x N] row-major into per-lane WMMA B-fragments (bf16).
// Tile (kt,nt): lane l -> col n = nt*16+(l&15), K base = kt*32+(l>>4)*8,
// elements 0..7 = K base+0..7, elements 8..15 = K base+16..23 (one v16bf per lane).
// ---------------------------------------------------------------------------
__global__ void pack_w(const float* __restrict__ W, __bf16* __restrict__ Wp, int K, int N) {
    int t = blockIdx.x * blockDim.x + threadIdx.x;
    int ntiles = N >> 4;
    int total  = (K >> 5) * ntiles * 32;
    if (t >= total) return;
    int lane = t & 31;
    int tile = t >> 5;
    int nt = tile % ntiles;
    int kt = tile / ntiles;
    int n = nt * 16 + (lane & 15);
    int kbase = kt * 32 + (lane >> 4) * 8;
    __bf16* o = Wp + (size_t)t * 16;
#pragma unroll
    for (int j = 0; j < 16; ++j)
        o[j] = (__bf16)W[(size_t)(kbase + j + (j >= 8 ? 8 : 0)) * N + n];
}

// ---------------------------------------------------------------------------
// out = act(h @ Wself + (agg*invdeg) @ Wneigh + b), K=128 fixed.
// One wave -> one 16x16 tile. blockDim = (N/16)*32. grid.x = M/16.
// ---------------------------------------------------------------------------
__device__ __forceinline__ v16bf load_a_frag(const float* base, float scale) {
    const float4* p = (const float4*)base;
    float4 f0 = p[0], f1 = p[1];   // K = kb .. kb+7
    float4 f2 = p[4], f3 = p[5];   // K = kb+16 .. kb+23
    v16bf a;
    a[0]  = (__bf16)(f0.x * scale); a[1]  = (__bf16)(f0.y * scale);
    a[2]  = (__bf16)(f0.z * scale); a[3]  = (__bf16)(f0.w * scale);
    a[4]  = (__bf16)(f1.x * scale); a[5]  = (__bf16)(f1.y * scale);
    a[6]  = (__bf16)(f1.z * scale); a[7]  = (__bf16)(f1.w * scale);
    a[8]  = (__bf16)(f2.x * scale); a[9]  = (__bf16)(f2.y * scale);
    a[10] = (__bf16)(f2.z * scale); a[11] = (__bf16)(f2.w * scale);
    a[12] = (__bf16)(f3.x * scale); a[13] = (__bf16)(f3.y * scale);
    a[14] = (__bf16)(f3.z * scale); a[15] = (__bf16)(f3.w * scale);
    return a;
}

__global__ void sage_gemm(const float* __restrict__ h,
                          const float* __restrict__ agg,
                          const float* __restrict__ invdeg,
                          const __bf16* __restrict__ Wself,
                          const __bf16* __restrict__ Wneigh,
                          const float* __restrict__ bias,
                          float* __restrict__ out,
                          int N, int relu_flag) {
    const int lane = threadIdx.x & 31;
    const int wave = threadIdx.x >> 5;           // = output column tile
    const int ntiles = N >> 4;
    const int row0 = blockIdx.x * 16;
    const int r  = row0 + (lane & 15);           // A row owned by this lane
    const int ko = (lane >> 4) * 8;              // per-lane K offset in A frag
    const int n  = wave * 16 + (lane & 15);      // output column

    const float s = invdeg[r];
    const float* hr = h   + (size_t)r * FEATS;
    const float* ar = agg + (size_t)r * FEATS;

    v8f acc = {};
#pragma unroll
    for (int kt = 0; kt < 4; ++kt) {
        const int kb = kt * 32 + ko;
        size_t widx = ((size_t)(kt * ntiles + wave) * 32 + lane) * 16;

        v16bf a_s = load_a_frag(hr + kb, 1.0f);
        v16bf b_s = *(const v16bf*)(Wself + widx);
        acc = __builtin_amdgcn_wmma_f32_16x16x32_bf16(
            false, a_s, false, b_s, (short)0, acc, false, false);

        v16bf a_n = load_a_frag(ar + kb, s);
        v16bf b_n = *(const v16bf*)(Wneigh + widx);
        acc = __builtin_amdgcn_wmma_f32_16x16x32_bf16(
            false, a_n, false, b_n, (short)0, acc, false, false);
    }

    const float bv = bias[n];
    const int mrow = row0 + (lane >> 4) * 8;
#pragma unroll
    for (int i = 0; i < 8; ++i) {
        float v = acc[i] + bv;
        if (relu_flag) v = fmaxf(v, 0.0f);
        out[(size_t)(mrow + i) * N + n] = v;
    }
}

// ---------------------------------------------------------------------------
// In-place log_softmax over 64 classes; one wave32 per row, 2 cols/lane.
// ---------------------------------------------------------------------------
__global__ void logsoftmax64(float* __restrict__ x, int n) {
    int lane = threadIdx.x & 31;
    int row  = blockIdx.x * (blockDim.x >> 5) + (threadIdx.x >> 5);
    if (row >= n) return;
    float* p = x + (size_t)row * 64;
    float v0 = p[lane], v1 = p[lane + 32];
    float m = fmaxf(v0, v1);
#pragma unroll
    for (int off = 16; off; off >>= 1) m = fmaxf(m, __shfl_xor(m, off, 32));
    float e = expf(v0 - m) + expf(v1 - m);
#pragma unroll
    for (int off = 16; off; off >>= 1) e += __shfl_xor(e, off, 32);
    float ls = logf(e);
    p[lane]      = v0 - m - ls;
    p[lane + 32] = v1 - m - ls;
}

// ---------------------------------------------------------------------------
extern "C" void kernel_launch(void* const* d_in, const int* in_sizes, int n_in,
                              void* d_out, int out_size, void* d_ws, size_t ws_size,
                              hipStream_t stream) {
    (void)in_sizes; (void)n_in; (void)out_size; (void)ws_size;
    const float* x   = (const float*)d_in[0];
    const int*   src = (const int*)  d_in[1];
    const int*   dst = (const int*)  d_in[2];
    const float* Ws0 = (const float*)d_in[3];
    const float* Wn0 = (const float*)d_in[4];
    const float* b0  = (const float*)d_in[5];
    const float* Ws1 = (const float*)d_in[6];
    const float* Wn1 = (const float*)d_in[7];
    const float* b1  = (const float*)d_in[8];
    const float* Ws2 = (const float*)d_in[9];
    const float* Wn2 = (const float*)d_in[10];
    const float* b2  = (const float*)d_in[11];

    // Workspace layout
    char* ws = (char*)d_ws;
    size_t off = 0;
    auto carve = [&](size_t bytes) -> void* {
        void* p = ws + off;
        off = (off + bytes + 255) & ~(size_t)255;
        return p;
    };
    float*  invdeg = (float*) carve((size_t)N_NODES * 4);
    float*  agg    = (float*) carve((size_t)N_NODES * FEATS * 4);
    float*  hA     = (float*) carve((size_t)N_NODES * FEATS * 4);
    float*  hB     = (float*) carve((size_t)N_NODES * FEATS * 4);
    __bf16* Wp_s0  = (__bf16*)carve(128 * 128 * 2);
    __bf16* Wp_n0  = (__bf16*)carve(128 * 128 * 2);
    __bf16* Wp_s1  = (__bf16*)carve(128 * 128 * 2);
    __bf16* Wp_n1  = (__bf16*)carve(128 * 128 * 2);
    __bf16* Wp_s2  = (__bf16*)carve(128 * 64 * 2);
    __bf16* Wp_n2  = (__bf16*)carve(128 * 64 * 2);

    // Degree -> inverse degree
    hipMemsetAsync(invdeg, 0, (size_t)N_NODES * 4, stream);
    deg_kernel<<<(N_EDGES + 255) / 256, 256, 0, stream>>>(dst, invdeg, N_EDGES);
    inv_kernel<<<(N_NODES + 255) / 256, 256, 0, stream>>>(invdeg, N_NODES);

    // Pack weights into WMMA B-fragment layout (bf16)
    pack_w<<<4, 256, 0, stream>>>(Ws0, Wp_s0, 128, 128);   // 1024 threads
    pack_w<<<4, 256, 0, stream>>>(Wn0, Wp_n0, 128, 128);
    pack_w<<<4, 256, 0, stream>>>(Ws1, Wp_s1, 128, 128);
    pack_w<<<4, 256, 0, stream>>>(Wn1, Wp_n1, 128, 128);
    pack_w<<<2, 256, 0, stream>>>(Ws2, Wp_s2, 128, 64);    // 512 threads
    pack_w<<<2, 256, 0, stream>>>(Wn2, Wp_n2, 128, 64);

    const int scatter_blocks = (int)(((size_t)N_EDGES * 32 + 255) / 256);
    const int gemm_blocks = N_NODES / 16;   // 6250, exact

    // Layer 0: x -> hA (relu)
    hipMemsetAsync(agg, 0, (size_t)N_NODES * FEATS * 4, stream);
    scatter_add<<<scatter_blocks, 256, 0, stream>>>(x, src, dst, agg, N_EDGES);
    sage_gemm<<<gemm_blocks, 256, 0, stream>>>(x, agg, invdeg, Wp_s0, Wp_n0, b0, hA, 128, 1);

    // Layer 1: hA -> hB (relu)
    hipMemsetAsync(agg, 0, (size_t)N_NODES * FEATS * 4, stream);
    scatter_add<<<scatter_blocks, 256, 0, stream>>>(hA, src, dst, agg, N_EDGES);
    sage_gemm<<<gemm_blocks, 256, 0, stream>>>(hA, agg, invdeg, Wp_s1, Wp_n1, b1, hB, 128, 1);

    // Layer 2: hB -> d_out (no relu), then log_softmax in place
    hipMemsetAsync(agg, 0, (size_t)N_NODES * FEATS * 4, stream);
    scatter_add<<<scatter_blocks, 256, 0, stream>>>(hB, src, dst, agg, N_EDGES);
    sage_gemm<<<gemm_blocks, 128, 0, stream>>>(hB, agg, invdeg, Wp_s2, Wp_n2, b2,
                                               (float*)d_out, 64, 0);
    logsoftmax64<<<(N_NODES + 7) / 8, 256, 0, stream>>>((float*)d_out, N_NODES);
}
